// FMRefinerTransformer_18236431139376
// MI455X (gfx1250) — matrix operand order repaired and checked by reading, not compiled, not run on hardware
//
#include <hip/hip_runtime.h>

// ---------------- problem constants ----------------
#define Jn 27
#define JP 32          // padded joints (tile-aligned)
#define Bn 64
#define Nn 2048
#define Dn 512
#define Hn 8
#define Ln 6
#define DFFn 2048
#define Rn (Bn*Jn)     // 1728 real rows
#define Rp (Bn*JP)     // 2048 padded rows of the residual stream (16 x 128)
#define BNn (Bn*Nn)    // 131072 radar tokens
#define NEGB (-1.0e9f)

typedef __bf16 bf16;
typedef __attribute__((ext_vector_type(16))) __bf16 v16bf;
typedef __attribute__((ext_vector_type(8)))  float   v8f;

__device__ __constant__ int c_parent[Jn] = {0,0,1,2,3,4,5,6,7,8,8,3,11,12,13,14,15,15,0,18,19,20,0,22,23,24,3};
__device__ __constant__ float c_scale[Jn] = {
  0.6f,0.6f,0.6f,0.6f, 1.f,1.f,1.f,1.f,1.f, 1.5f,1.5f, 1.f,1.f,1.f,1.f,1.f,
  1.5f,1.5f, 0.6f, 1.f, 1.5f,1.5f, 0.6f, 1.f, 1.5f,1.5f,1.5f };

static __device__ __forceinline__ float dev_silu(float v){ return v * (1.0f/(1.0f + __expf(-v))); }
static inline int cdiv(int a,int b){ return (a+b-1)/b; }

// ---------------- generic bf16 WMMA GEMM (branch-free inner loop) ----------------
// C[z](M,N) = act( alpha * A[z](M,K) * B[z](K,N) + bias + resid )
// A row-major (lda). B col-major: element (k,n) at B[n*ldb + k].
// Launch invariants (guaranteed by all call sites): K % 32 == 0; every wave tile
// that passes the uniform early-return is fully in-bounds (M % 32 == 0 tiles,
// N % 64 == 0 tiles). No per-element guards needed.
struct Gemm {
  const bf16* A; const bf16* Bm; const float* bias; const float* resid; void* C;
  long long aO, aI, bO, bI, biasI, cO, cI;
  int inner, lda, ldb, ldc, M, N, K;
  float alpha; int act, cBf16;
};

template<int WM, int WN>
__global__ __launch_bounds__(WM*WN*32)
void k_gemm(Gemm g) {
  const int lane = threadIdx.x & 31;
  const int wave = threadIdx.x >> 5;
  const int wm = wave % WM, wn = wave / WM;
  const int zo = blockIdx.z / g.inner, zi = blockIdx.z - zo*g.inner;
  const int row0 = (blockIdx.y*WM + wm) * 32;
  const int col0 = (blockIdx.x*WN + wn) * 64;
  if (row0 >= g.M || col0 >= g.N) return;           // wave-uniform; no barriers used
  const bf16* Ap = g.A + (long long)zo*g.aO + (long long)zi*g.aI;
  const bf16* Bp = g.Bm + (long long)zo*g.bO + (long long)zi*g.bI;
  const int hi = (lane >> 4) & 1;                   // half-wave select
  const bf16* pa  = Ap + (long long)(row0 + (lane & 15))*g.lda + hi*16;
  const bf16* pb0 = Bp + (long long)(col0 + (lane & 15))*g.ldb + hi*16;
  const long long aRow16 = (long long)16 * g.lda;
  const long long bCol16 = (long long)16 * g.ldb;

  v8f acc[2][4];
  #pragma unroll
  for (int a=0;a<2;++a)
    #pragma unroll
    for (int c=0;c<4;++c)
      #pragma unroll
      for (int i=0;i<8;++i) acc[a][c][i] = 0.0f;

  for (int kb = 0; kb < g.K; kb += 32) {
    v16bf af[2];
    #pragma unroll
    for (int mi = 0; mi < 2; ++mi) {
      const bf16* p = pa + (long long)mi*aRow16 + kb;
      #pragma unroll
      for (int i=0;i<16;++i) af[mi][i] = p[i];                             // contiguous 32B
    }
    v16bf bfr[4];
    #pragma unroll
    for (int ni = 0; ni < 4; ++ni) {
      const bf16* p = pb0 + (long long)ni*bCol16 + kb;
      #pragma unroll
      for (int i=0;i<16;++i) bfr[ni][i] = p[i];                            // contiguous 32B
    }
    #pragma unroll
    for (int mi=0;mi<2;++mi)
      #pragma unroll
      for (int ni=0;ni<4;++ni)
        acc[mi][ni] = __builtin_amdgcn_wmma_f32_16x16x32_bf16(
            false, af[mi], false, bfr[ni], (short)0, acc[mi][ni], false, false);
    if (kb + 32 < g.K) {
      __builtin_prefetch(pa + kb + 32, 0, 3);        // near-cache prefetch of next A line
      __builtin_prefetch(pb0 + kb + 32, 0, 3);
    }
  }

  const float* bp = g.bias ? (g.bias + (long long)zi*g.biasI) : nullptr;
  #pragma unroll
  for (int mi=0;mi<2;++mi) {
    int rb = row0 + mi*16 + hi*8;
    #pragma unroll
    for (int ni=0;ni<4;++ni) {
      int c = col0 + ni*16 + (lane & 15);
      float bb = bp ? bp[c] : 0.0f;
      #pragma unroll
      for (int i=0;i<8;++i) {
        int r = rb + i;
        float v = acc[mi][ni][i] * g.alpha + bb;
        if (g.resid) v += g.resid[(long long)r*g.ldc + c];
        if (g.act == 1) v = fmaxf(v, 0.0f);
        else if (g.act == 2) v = dev_silu(v);
        long long ci = (long long)zo*g.cO + (long long)zi*g.cI + (long long)r*g.ldc + c;
        if (g.cBf16) ((bf16*)g.C)[ci] = (bf16)v;
        else         ((float*)g.C)[ci] = v;
      }
    }
  }
}

// ---------------- conversion / transpose ----------------
__global__ void k_cvt(const float* __restrict__ s, bf16* __restrict__ d, long long n) {
  long long i = (long long)blockIdx.x*256 + threadIdx.x;
  if (i < n) d[i] = (bf16)s[i];
}

template<typename T>
__global__ void k_T(const T* __restrict__ src, bf16* __restrict__ dst, int Rr, int Cc) {
  __shared__ float tile[32][33];
  long long zb = (long long)blockIdx.z * Rr * Cc;
  int r0 = blockIdx.y*32, c0 = blockIdx.x*32;
  for (int i = threadIdx.y; i < 32; i += 8) {
    int r = r0 + i, c = c0 + threadIdx.x;
    tile[i][threadIdx.x] = (r < Rr && c < Cc) ? (float)src[zb + (long long)r*Cc + c] : 0.0f;
  }
  __syncthreads();
  for (int i = threadIdx.y; i < 32; i += 8) {
    int c = c0 + i, r = r0 + threadIdx.x;
    if (r < Rr && c < Cc) dst[zb + (long long)c*Rr + r] = (bf16)tile[threadIdx.x][i];
  }
}

// ---------------- preprocessing kernels ----------------
// per real (b,j): dw row, sum(+1e-6), argmax(dw)=argmin(dist), all-outside flag.
// denom/minidx/allout are written at PADDED index b*JP+j.
__global__ void k_dw(const float* __restrict__ xt, const float* __restrict__ pc,
                     float* __restrict__ dw, float* __restrict__ denom,
                     int* __restrict__ minidx, int* __restrict__ allout) {
  __shared__ float rs[256]; __shared__ float rw[256]; __shared__ int ri[256];
  int bj = blockIdx.x; int b = bj / Jn, j = bj - b*Jn;
  float x0 = xt[bj*3], x1 = xt[bj*3+1], x2 = xt[bj*3+2];
  float ar = c_scale[j] * 0.1f;
  float coef = 2.0f / (ar*ar);
  float sum = 0.0f, best = -1.0f; int bi = Nn;
  for (int n = threadIdx.x; n < Nn; n += 256) {
    const float* p = pc + ((long long)b*Nn + n)*6;
    float dx = x0 - p[0], dy = x1 - p[1], dz = x2 - p[2];
    float d2 = dx*dx + dy*dy + dz*dz;
    float w = __expf(-d2 * coef);
    dw[(long long)bj*Nn + n] = w;
    sum += w;
    if (w > best) { best = w; bi = n; }
  }
  int t = threadIdx.x;
  rs[t] = sum; rw[t] = best; ri[t] = bi; __syncthreads();
  for (int o = 128; o > 0; o >>= 1) {
    if (t < o) {
      rs[t] += rs[t+o];
      if (rw[t+o] > rw[t] || (rw[t+o] == rw[t] && ri[t+o] < ri[t])) { rw[t] = rw[t+o]; ri[t] = ri[t+o]; }
    }
    __syncthreads();
  }
  if (t == 0) {
    int bjp = b*JP + j;
    denom[bjp]  = rs[0] + 1e-6f;
    minidx[bjp] = ri[0];
    allout[bjp] = (rw[0] < 0.13533528f) ? 1 : 0;   // max dw < e^-2  <=>  min dist > ar
  }
}

__global__ void k_pi(const float* __restrict__ dw, float* __restrict__ pi) {
  int i = blockIdx.x*256 + threadIdx.x; if (i >= BNn) return;
  int b = i / Nn, n = i - b*Nn;
  const float* p = dw + (long long)b*Jn*Nn + n;
  float m = p[0];
  for (int j = 1; j < Jn; ++j) m = fmaxf(m, p[(long long)j*Nn]);
  pi[i] = m;
}

// membias(b,j,n) (27-row layout) + padded bf16 copy of dw (GEMM A operand, JP rows)
__global__ void k_membias(const float* __restrict__ dw, const int* __restrict__ minidx,
                          const int* __restrict__ allout, float* __restrict__ mb,
                          bf16* __restrict__ dwb) {
  long long i = (long long)blockIdx.x*256 + threadIdx.x;
  if (i >= (long long)Rn*Nn) return;
  int n = (int)(i % Nn); int bj = (int)(i / Nn);
  int b = bj / Jn, j = bj - b*Jn;
  int bjp = b*JP + j;
  float w = dw[i];
  dwb[((long long)bjp)*Nn + n] = (bf16)w;
  float v;
  if (allout[bjp]) v = (n == minidx[bjp]) ? 0.0f : NEGB;
  else             v = (w < 0.13533528f) ? NEGB : 0.0f;
  mb[i] = v;
}

// memory = (radar_pc @ pc_w + pc_b) * pi  (bf16, single HBM pass; L2-resident afterwards)
__global__ void k_mem(const float* __restrict__ pc, const float* __restrict__ w,
                      const float* __restrict__ bb, const float* __restrict__ pi,
                      bf16* __restrict__ mem) {
  long long i = (long long)blockIdx.x*256 + threadIdx.x;
  if (i >= (long long)BNn*Dn) return;
  int d = (int)(i % Dn); long long bn = i / Dn;
  const float* p = pc + bn*6;
  float s = bb[d];
  #pragma unroll
  for (int c = 0; c < 6; ++c) s += p[c] * w[c*Dn + d];
  mem[i] = (bf16)(s * pi[bn]);
}

// hidden = silu(dop @ dop_w1 + dop_b1)  (bf16, 64-wide)
__global__ void k_dophid(const float* __restrict__ pc, const float* __restrict__ w1,
                         const float* __restrict__ b1, bf16* __restrict__ hid) {
  long long bn = (long long)blockIdx.x*256 + threadIdx.x;
  if (bn >= BNn) return;
  float dop = pc[bn*6 + 3];
  bf16* o = hid + bn*64;
  for (int j = 0; j < 64; ++j) o[j] = (bf16)dev_silu(dop * w1[j] + b1[j]);
}

// h_state / h_coarse / concat(h_state, h_parent); pad rows (j>=27) get zeros.
__global__ void k_smallfeats(const float* __restrict__ xt, const float* __restrict__ cp,
                             const float* __restrict__ sw, const float* __restrict__ sb,
                             const float* __restrict__ cw, const float* __restrict__ cb,
                             float* __restrict__ hstate, float* __restrict__ hcoarse,
                             bf16* __restrict__ concat) {
  long long i = (long long)blockIdx.x*256 + threadIdx.x;
  if (i >= (long long)Rp*Dn) return;
  int d = (int)(i % Dn); int row = (int)(i / Dn);
  int b = row / JP, j = row - b*JP;
  if (j >= Jn) {
    hstate[i] = 0.0f; hcoarse[i] = 0.0f;
    concat[(long long)row*1024 + d]      = (bf16)0.0f;
    concat[(long long)row*1024 + Dn + d] = (bf16)0.0f;
    return;
  }
  int jp = c_parent[j];
  const float* x0 = xt + ((long long)b*Jn + j)*3;
  const float* xp = xt + ((long long)b*Jn + jp)*3;
  const float* c0 = cp + ((long long)b*Jn + j)*3;
  float hs = sb[d], hp = sb[d], hc = cb[d];
  #pragma unroll
  for (int c = 0; c < 3; ++c) {
    hs += x0[c] * sw[c*Dn + d];
    hp += xp[c] * sw[c*Dn + d];
    hc += c0[c] * cw[c*Dn + d];
  }
  hstate[i] = hs; hcoarse[i] = hc;
  concat[(long long)row*1024 + d]      = (bf16)hs;
  concat[(long long)row*1024 + Dn + d] = (bf16)hp;
}

__global__ void k_time1(const float* __restrict__ t, const float* __restrict__ w1,
                        const float* __restrict__ b1, float* __restrict__ h) {
  int i = blockIdx.x*256 + threadIdx.x; if (i >= Bn*Dn) return;
  int b = i / Dn, d = i - b*Dn;
  h[i] = dev_silu(t[b] * w1[d] + b1[d]);
}
__global__ void k_time2(const float* __restrict__ h1, const float* __restrict__ w2,
                        const float* __restrict__ b2, float* __restrict__ ht) {
  int i = blockIdx.x*256 + threadIdx.x; if (i >= Bn*Dn) return;
  int b = i / Dn, d = i - b*Dn;
  float s = b2[d];
  const float* hr = h1 + (long long)b*Dn;
  for (int e = 0; e < Dn; ++e) s += hr[e] * w2[e*Dn + d];
  ht[i] = s;
}

// query assembly + AdaLN modulation -> residual stream x (fp32, JP-padded rows)
// h_dop bias correction: dop_feat carried +dop_b2, so add (sum dw)*b2 = (denom-1e-6)*b2.
__global__ void k_query(const float* __restrict__ hcoarse, const float* __restrict__ htime,
                        const float* __restrict__ hdop, const float* __restrict__ denom,
                        const float* __restrict__ dopb2,
                        const float* __restrict__ hstate, const float* __restrict__ hstruct,
                        const float* __restrict__ jid, const float* __restrict__ modf,
                        float* __restrict__ x) {
  long long i = (long long)blockIdx.x*256 + threadIdx.x;
  if (i >= (long long)Rp*Dn) return;
  int d = (int)(i % Dn); int row = (int)(i / Dn);
  int b = row / JP, j = row - b*JP;
  if (j >= Jn) { x[i] = 0.0f; return; }
  float dn = denom[row];
  float hd = (hdop[i] + (dn - 1e-6f) * dopb2[d]) / dn;
  float q = hcoarse[i] + htime[(long long)b*Dn + d] + hd
          + hstate[i] + hstruct[i] + jid[(long long)j*Dn + d];
  float sh = modf[(long long)row*1024 + d];
  float sc = modf[(long long)row*1024 + Dn + d];
  x[i] = q * (1.0f + sc) + sh;
}

// ---------------- transformer kernels ----------------
__global__ void k_ln(const float* __restrict__ x, const float* __restrict__ s,
                     const float* __restrict__ bb, bf16* __restrict__ y) {
  __shared__ float red[256];
  int row = blockIdx.x, t = threadIdx.x;
  const float* xr = x + (long long)row*Dn;
  float v0 = xr[t], v1 = xr[t+256];
  red[t] = v0 + v1; __syncthreads();
  for (int o = 128; o > 0; o >>= 1) { if (t < o) red[t] += red[t+o]; __syncthreads(); }
  float mu = red[0] * (1.0f/Dn); __syncthreads();
  float d0 = v0 - mu, d1 = v1 - mu;
  red[t] = d0*d0 + d1*d1; __syncthreads();
  for (int o = 128; o > 0; o >>= 1) { if (t < o) red[t] += red[t+o]; __syncthreads(); }
  float rs = rsqrtf(red[0] * (1.0f/Dn) + 1e-5f);
  bf16* yr = y + (long long)row*Dn;
  yr[t]     = (bf16)(d0*rs*s[t]     + bb[t]);
  yr[t+256] = (bf16)(d1*rs*s[t+256] + bb[t+256]);
}

// self attention (27x27 per (b,h)), topo adjacency mask, fp32 VALU; rows JP-padded
__global__ void k_selfattn(const float* __restrict__ qkv, bf16* __restrict__ o) {
  int bh = blockIdx.x; int b = bh >> 3, h = bh & 7;
  int j = threadIdx.x;
  if (j >= Jn) return;
  const float* base = qkv + (long long)b*JP*1536;
  float qv[64];
  const float* qj = base + (long long)j*1536 + h*64;
  #pragma unroll
  for (int d = 0; d < 64; ++d) qv[d] = qj[d];
  float sc[Jn]; float mx = -3.0e38f;
  for (int kk = 0; kk < Jn; ++kk) {
    const float* kr = base + (long long)kk*1536 + Dn + h*64;
    float s = 0.0f;
    #pragma unroll
    for (int d = 0; d < 64; ++d) s += qv[d] * kr[d];
    s *= 0.125f;
    bool adj = (kk == j) || (c_parent[j] == kk) || (c_parent[kk] == j);
    if (!adj) s += NEGB;
    sc[kk] = s; mx = fmaxf(mx, s);
  }
  float den = 0.0f;
  for (int kk = 0; kk < Jn; ++kk) { sc[kk] = __expf(sc[kk] - mx); den += sc[kk]; }
  float inv = 1.0f / den;
  float ov[64];
  #pragma unroll
  for (int d = 0; d < 64; ++d) ov[d] = 0.0f;
  for (int kk = 0; kk < Jn; ++kk) {
    float wgt = sc[kk] * inv;
    const float* vr = base + (long long)kk*1536 + 2*Dn + h*64;
    #pragma unroll
    for (int d = 0; d < 64; ++d) ov[d] += wgt * vr[d];
  }
  bf16* orow = o + ((long long)(b*JP + j))*Dn + h*64;
  #pragma unroll
  for (int d = 0; d < 64; ++d) orow[d] = (bf16)ov[d];
}

// softmax over 2048 keys with memory bias; scores/attn layout (B,H,JP,N), real q only
__global__ void k_softmax(const float* __restrict__ sc, const float* __restrict__ mb,
                          bf16* __restrict__ attn) {
  __shared__ float red[256];
  int id = blockIdx.x;                        // (b*H + h)*27 + q
  int q = id % Jn; int bh = id / Jn; int b = bh / Hn;
  long long row = (long long)bh*JP + q;
  const float* sr = sc + row*Nn;
  const float* mr = mb + ((long long)(b*Jn + q))*Nn;
  int t = threadIdx.x;
  float v[8]; float mx = -3.0e38f;
  #pragma unroll
  for (int i = 0; i < 8; ++i) { int n = t + i*256; v[i] = sr[n] + mr[n]; mx = fmaxf(mx, v[i]); }
  red[t] = mx; __syncthreads();
  for (int o = 128; o > 0; o >>= 1) { if (t < o) red[t] = fmaxf(red[t], red[t+o]); __syncthreads(); }
  mx = red[0]; __syncthreads();
  float s = 0.0f;
  #pragma unroll
  for (int i = 0; i < 8; ++i) { v[i] = __expf(v[i] - mx); s += v[i]; }
  red[t] = s; __syncthreads();
  for (int o = 128; o > 0; o >>= 1) { if (t < o) red[t] += red[t+o]; __syncthreads(); }
  float inv = 1.0f / red[0];
  bf16* ar = attn + row*Nn;
  #pragma unroll
  for (int i = 0; i < 8; ++i) ar[t + i*256] = (bf16)(v[i]*inv);
}

__global__ void k_vel(const float* __restrict__ x, const float* __restrict__ w,
                      const float* __restrict__ bb, float* __restrict__ out) {
  int i = blockIdx.x*256 + threadIdx.x;
  if (i >= Rn*3) return;
  int rr = i / 3, c = i - rr*3;
  int b = rr / Jn, j = rr - b*Jn;
  float s = bb[c];
  const float* xr = x + ((long long)b*JP + j)*Dn;
  for (int e = 0; e < Dn; ++e) s += xr[e] * w[e*3 + c];
  out[i] = s;
}

// ---------------- host orchestration ----------------
static void gemm(hipStream_t st, int M, int N, int K,
                 const bf16* A, int lda, long long aO, long long aI,
                 const bf16* Bm, int ldb, long long bO, long long bI,
                 const float* bias, long long biasI,
                 const float* resid,
                 void* C, int ldc, int cBf16, long long cO, long long cI,
                 int batches, int inner, float alpha, int act)
{
  Gemm g;
  g.A = A; g.Bm = Bm; g.bias = bias; g.resid = resid; g.C = C;
  g.aO = aO; g.aI = aI; g.bO = bO; g.bI = bI; g.biasI = biasI; g.cO = cO; g.cI = cI;
  g.inner = inner; g.lda = lda; g.ldb = ldb; g.ldc = ldc; g.M = M; g.N = N; g.K = K;
  g.alpha = alpha; g.act = act; g.cBf16 = cBf16;
  if (M <= 32) {
    dim3 blk(64), gr(cdiv(N,128), 1, batches);
    k_gemm<1,2><<<gr, blk, 0, st>>>(g);
  } else {
    dim3 blk(256), gr(cdiv(N,128), cdiv(M,128), batches);
    k_gemm<4,2><<<gr, blk, 0, st>>>(g);
  }
}

extern "C" void kernel_launch(void* const* d_in, const int* in_sizes, int n_in,
                              void* d_out, int out_size, void* d_ws, size_t ws_size,
                              hipStream_t stream) {
  (void)in_sizes; (void)n_in; (void)out_size; (void)ws_size;
  const float* x_t       = (const float*)d_in[0];
  const float* t_in      = (const float*)d_in[1];
  const float* coarse    = (const float*)d_in[2];
  const float* radar     = (const float*)d_in[3];
  const float* pc_w      = (const float*)d_in[4];
  const float* pc_b      = (const float*)d_in[5];
  const float* coarse_w  = (const float*)d_in[6];
  const float* coarse_b  = (const float*)d_in[7];
  const float* state_w   = (const float*)d_in[8];
  const float* state_b   = (const float*)d_in[9];
  const float* jid_emb   = (const float*)d_in[10];
  const float* dop_w1    = (const float*)d_in[11];
  const float* dop_b1    = (const float*)d_in[12];
  const float* dop_w2    = (const float*)d_in[13];
  const float* dop_b2    = (const float*)d_in[14];
  const float* time_w1   = (const float*)d_in[15];
  const float* time_b1   = (const float*)d_in[16];
  const float* time_w2   = (const float*)d_in[17];
  const float* time_b2   = (const float*)d_in[18];
  const float* mod_w1    = (const float*)d_in[19];
  const float* mod_b1    = (const float*)d_in[20];
  const float* mod_w2    = (const float*)d_in[21];
  const float* mod_b2    = (const float*)d_in[22];
  const float* diff_w    = (const float*)d_in[23];
  const float* diff_b    = (const float*)d_in[24];
  const float* vel_w     = (const float*)d_in[25];
  const float* vel_b     = (const float*)d_in[26];
  const float* sa_qkv_w  = (const float*)d_in[27];
  const float* sa_qkv_b  = (const float*)d_in[28];
  const float* sa_out_w  = (const float*)d_in[29];
  const float* sa_out_b  = (const float*)d_in[30];
  const float* ca_qkv_w  = (const float*)d_in[31];
  const float* ca_qkv_b  = (const float*)d_in[32];
  const float* ca_out_w  = (const float*)d_in[33];
  const float* ca_out_b  = (const float*)d_in[34];
  const float* ff1_w     = (const float*)d_in[35];
  const float* ff1_b     = (const float*)d_in[36];
  const float* ff2_w     = (const float*)d_in[37];
  const float* ff2_b     = (const float*)d_in[38];
  const float* ln1_s     = (const float*)d_in[39];
  const float* ln1_b     = (const float*)d_in[40];
  const float* ln2_s     = (const float*)d_in[41];
  const float* ln2_b     = (const float*)d_in[42];
  const float* ln3_s     = (const float*)d_in[43];
  const float* ln3_b     = (const float*)d_in[44];

  // ---- workspace arena (deterministic layout every call) ----
  char* base = (char*)d_ws;
  size_t off = 0;
  auto AL = [&](size_t bytes) -> char* {
    size_t a = (off + 255) & ~(size_t)255; off = a + bytes; return base + a;
  };
  // bf16 weights
  bf16* w_saqkvT = (bf16*)AL((size_t)Ln*1536*512*2);
  bf16* w_saoutT = (bf16*)AL((size_t)Ln*512*512*2);
  bf16* w_caqkvT = (bf16*)AL((size_t)Ln*1536*512*2);
  bf16* w_caqkv  = (bf16*)AL((size_t)Ln*512*1536*2);
  bf16* w_caoutT = (bf16*)AL((size_t)Ln*512*512*2);
  bf16* w_ff1T   = (bf16*)AL((size_t)Ln*2048*512*2);
  bf16* w_ff2T   = (bf16*)AL((size_t)Ln*512*2048*2);
  bf16* w_mod1T  = (bf16*)AL((size_t)1024*512*2);
  bf16* w_mod2T  = (bf16*)AL((size_t)1024*1024*2);
  bf16* w_diffT  = (bf16*)AL((size_t)512*1024*2);
  bf16* w_dop2T  = (bf16*)AL((size_t)512*64*2);
  // activations
  bf16* mem_bf   = (bf16*)AL((size_t)BNn*Dn*2);      // stays L2-resident across layers
  bf16* memT_bf  = (bf16*)AL((size_t)BNn*Dn*2);
  bf16* dop_hid  = (bf16*)AL((size_t)BNn*64*2);
  bf16* hidT     = (bf16*)AL((size_t)BNn*64*2);      // (B,64,N)
  bf16* hdop64   = (bf16*)AL((size_t)Bn*JP*64*2);
  bf16* dw_bf    = (bf16*)AL((size_t)Bn*JP*Nn*2);    // JP-padded A operand
  bf16* concat_b = (bf16*)AL((size_t)Rp*1024*2);
  bf16* modh_b   = (bf16*)AL((size_t)Rp*1024*2);
  bf16* hln      = (bf16*)AL((size_t)Rp*Dn*2);
  bf16* o_s      = (bf16*)AL((size_t)Rp*Dn*2);
  bf16* qbuf     = (bf16*)AL((size_t)Rp*Dn*2);
  bf16* QtB      = (bf16*)AL((size_t)Bn*Hn*JP*Dn*2); // reused for ctx
  bf16* attn     = (bf16*)AL((size_t)Bn*Hn*JP*Nn*2);
  bf16* o_asm    = (bf16*)AL((size_t)Rp*Dn*2);
  bf16* ffh      = (bf16*)AL((size_t)Rp*DFFn*2);
  float* dw_f    = (float*)AL((size_t)Rn*Nn*4);
  float* membias = (float*)AL((size_t)Rn*Nn*4);
  float* pi      = (float*)AL((size_t)BNn*4);
  float* denom   = (float*)AL((size_t)Bn*JP*4);
  float* hstate  = (float*)AL((size_t)Rp*Dn*4);
  float* hcoarse = (float*)AL((size_t)Rp*Dn*4);
  float* hstruct = (float*)AL((size_t)Rp*Dn*4);
  float* hdop    = (float*)AL((size_t)Rp*Dn*4);
  float* ht1     = (float*)AL((size_t)Bn*Dn*4);
  float* htime   = (float*)AL((size_t)Bn*Dn*4);
  float* modf    = (float*)AL((size_t)Rp*1024*4);
  float* xres    = (float*)AL((size_t)Rp*Dn*4);
  float* qkv_s   = (float*)AL((size_t)Rp*1536*4);
  float* scores  = (float*)AL((size_t)Bn*Hn*JP*Nn*4);
  int*   minidx  = (int*)AL((size_t)Bn*JP*4);
  int*   allout  = (int*)AL((size_t)Bn*JP*4);

  dim3 tb(32,8);
  // ---- weight conversion / transposition ----
  k_T<float><<<dim3(48,16,Ln), tb, 0, stream>>>(sa_qkv_w, w_saqkvT, 512, 1536);
  k_T<float><<<dim3(16,16,Ln), tb, 0, stream>>>(sa_out_w, w_saoutT, 512, 512);
  k_T<float><<<dim3(48,16,Ln), tb, 0, stream>>>(ca_qkv_w, w_caqkvT, 512, 1536);
  k_cvt<<<dim3(cdiv(Ln*512*1536,256)), dim3(256), 0, stream>>>(ca_qkv_w, w_caqkv, (long long)Ln*512*1536);
  k_T<float><<<dim3(16,16,Ln), tb, 0, stream>>>(ca_out_w, w_caoutT, 512, 512);
  k_T<float><<<dim3(64,16,Ln), tb, 0, stream>>>(ff1_w, w_ff1T, 512, 2048);
  k_T<float><<<dim3(16,64,Ln), tb, 0, stream>>>(ff2_w, w_ff2T, 2048, 512);
  k_T<float><<<dim3(32,16,1), tb, 0, stream>>>(mod_w1, w_mod1T, 512, 1024);
  k_T<float><<<dim3(32,32,1), tb, 0, stream>>>(mod_w2, w_mod2T, 1024, 1024);
  k_T<float><<<dim3(16,32,1), tb, 0, stream>>>(diff_w, w_diffT, 1024, 512);
  k_T<float><<<dim3(16,2,1),  tb, 0, stream>>>(dop_w2, w_dop2T, 64, 512);

  // ---- preprocessing ----
  k_dw<<<dim3(Rn), dim3(256), 0, stream>>>(x_t, radar, dw_f, denom, minidx, allout);
  k_pi<<<dim3(cdiv(BNn,256)), dim3(256), 0, stream>>>(dw_f, pi);
  k_membias<<<dim3(cdiv(Rn*Nn,256)), dim3(256), 0, stream>>>(dw_f, minidx, allout, membias, dw_bf);
  k_mem<<<dim3((int)(((long long)BNn*Dn + 255)/256)), dim3(256), 0, stream>>>(radar, pc_w, pc_b, pi, mem_bf);
  k_T<bf16><<<dim3(16,64,Bn), tb, 0, stream>>>(mem_bf, memT_bf, Nn, Dn);   // memT[b](D,N)
  k_dophid<<<dim3(cdiv(BNn,256)), dim3(256), 0, stream>>>(radar, dop_w1, dop_b1, dop_hid);
  k_T<bf16><<<dim3(2,64,Bn), tb, 0, stream>>>(dop_hid, hidT, Nn, 64);      // hidT[b](64,N)
  // hdop64[b](32,64) = dw(32,2048) @ hidden(2048,64)
  gemm(stream, JP, 64, Nn, dw_bf, Nn, (long long)JP*Nn, 0,
       hidT, Nn, (long long)64*Nn, 0, nullptr, 0, nullptr,
       hdop64, 64, 1, (long long)JP*64, 0, Bn, 1, 1.0f, 0);
  // hdop[b](32,512) = hdop64 @ dop_w2(64,512)   (dop_b2 handled in k_query)
  gemm(stream, JP, Dn, 64, hdop64, 64, (long long)JP*64, 0,
       w_dop2T, 64, 0, 0, nullptr, 0, nullptr,
       hdop, Dn, 0, (long long)JP*Dn, 0, Bn, 1, 1.0f, 0);
  k_smallfeats<<<dim3(cdiv(Rp*Dn,256)), dim3(256), 0, stream>>>(x_t, coarse, state_w, state_b,
                                                                coarse_w, coarse_b, hstate, hcoarse, concat_b);
  gemm(stream, Rp, Dn, 1024, concat_b, 1024, 0,0, w_diffT, 1024, 0,0,
       diff_b, 0, nullptr, hstruct, Dn, 0, 0,0, 1, 1, 1.0f, 0);
  gemm(stream, Rp, 1024, 512, concat_b, 1024, 0,0, w_mod1T, 512, 0,0,
       mod_b1, 0, nullptr, modh_b, 1024, 1, 0,0, 1, 1, 1.0f, 2);
  gemm(stream, Rp, 1024, 1024, modh_b, 1024, 0,0, w_mod2T, 1024, 0,0,
       mod_b2, 0, nullptr, modf, 1024, 0, 0,0, 1, 1, 1.0f, 0);
  k_time1<<<dim3(cdiv(Bn*Dn,256)), dim3(256), 0, stream>>>(t_in, time_w1, time_b1, ht1);
  k_time2<<<dim3(cdiv(Bn*Dn,256)), dim3(256), 0, stream>>>(ht1, time_w2, time_b2, htime);
  k_query<<<dim3(cdiv(Rp*Dn,256)), dim3(256), 0, stream>>>(hcoarse, htime, hdop, denom, dop_b2,
                                                           hstate, hstruct, jid_emb, modf, xres);

  // ---- transformer layers ----
  for (int l = 0; l < Ln; ++l) {
    // self attention block
    k_ln<<<dim3(Rp), dim3(256), 0, stream>>>(xres, ln1_s + l*Dn, ln1_b + l*Dn, hln);
    gemm(stream, Rp, 1536, 512, hln, Dn, 0,0, w_saqkvT + (size_t)l*1536*512, 512, 0,0,
         sa_qkv_b + l*1536, 0, nullptr, qkv_s, 1536, 0, 0,0, 1, 1, 1.0f, 0);
    k_selfattn<<<dim3(Bn*Hn), dim3(32), 0, stream>>>(qkv_s, o_s);
    gemm(stream, Rp, Dn, 512, o_s, Dn, 0,0, w_saoutT + (size_t)l*512*512, 512, 0,0,
         sa_out_b + l*Dn, 0, xres, xres, Dn, 0, 0,0, 1, 1, 1.0f, 0);

    // cross attention (K/V folded into query side; memory stays L2-resident)
    k_ln<<<dim3(Rp), dim3(256), 0, stream>>>(xres, ln2_s + l*Dn, ln2_b + l*Dn, hln);
    gemm(stream, Rp, Dn, 512, hln, Dn, 0,0, w_caqkvT + (size_t)l*1536*512, 512, 0,0,
         ca_qkv_b + l*1536, 0, nullptr, qbuf, Dn, 1, 0,0, 1, 1, 1.0f, 0);
    // Qt[b,h](32,512) = q_head(32,64) @ Wk_head^T
    gemm(stream, JP, Dn, 64, qbuf, Dn, (long long)JP*Dn, 64,
         w_caqkv + (size_t)l*512*1536 + 512, 1536, 0, 64,
         nullptr, 0, nullptr, QtB, Dn, 1, (long long)Hn*JP*Dn, (long long)JP*Dn,
         Bn*Hn, Hn, 1.0f, 0);
    // scores[b,h](32,2048) = Qt @ mem[b]^T / sqrt(64)
    gemm(stream, JP, Nn, Dn, QtB, Dn, (long long)Hn*JP*Dn, (long long)JP*Dn,
         mem_bf, Dn, (long long)Nn*Dn, 0,
         nullptr, 0, nullptr, scores, Nn, 0, (long long)Hn*JP*Nn, (long long)JP*Nn,
         Bn*Hn, Hn, 0.125f, 0);
    k_softmax<<<dim3(Bn*Hn*Jn), dim3(256), 0, stream>>>(scores, membias, attn);
    // ctx[b,h](32,512) = attn(32,2048) @ mem[b](2048,512)   (reuse QtB)
    gemm(stream, JP, Dn, Nn, attn, Nn, (long long)Hn*JP*Nn, (long long)JP*Nn,
         memT_bf, Nn, (long long)Dn*Nn, 0,
         nullptr, 0, nullptr, QtB, Dn, 1, (long long)Hn*JP*Dn, (long long)JP*Dn,
         Bn*Hn, Hn, 1.0f, 0);
    // o_head(32,64) = ctx(32,512) @ Wv_head(512,64) + bv
    gemm(stream, JP, 64, Dn, QtB, Dn, (long long)Hn*JP*Dn, (long long)JP*Dn,
         w_caqkvT + (size_t)l*1536*512 + (size_t)1024*512, 512, 0, (long long)64*512,
         ca_qkv_b + l*1536 + 1024, 64, nullptr, o_asm, Dn, 1, (long long)JP*Dn, 64,
         Bn*Hn, Hn, 1.0f, 0);
    gemm(stream, Rp, Dn, 512, o_asm, Dn, 0,0, w_caoutT + (size_t)l*512*512, 512, 0,0,
         ca_out_b + l*Dn, 0, xres, xres, Dn, 0, 0,0, 1, 1, 1.0f, 0);

    // FFN block
    k_ln<<<dim3(Rp), dim3(256), 0, stream>>>(xres, ln3_s + l*Dn, ln3_b + l*Dn, hln);
    gemm(stream, Rp, DFFn, 512, hln, Dn, 0,0, w_ff1T + (size_t)l*2048*512, 512, 0,0,
         ff1_b + l*DFFn, 0, nullptr, ffh, DFFn, 1, 0,0, 1, 1, 1.0f, 1);
    gemm(stream, Rp, Dn, DFFn, ffh, DFFn, 0,0, w_ff2T + (size_t)l*512*2048, 2048, 0,0,
         ff2_b + l*Dn, 0, xres, xres, Dn, 0, 0,0, 1, 1, 1.0f, 0);
  }

  k_vel<<<dim3(cdiv(Rn*3,256)), dim3(256), 0, stream>>>(xres, vel_w, vel_b, (float*)d_out);
}
